// MultiHeadAttention_8684423872844
// MI455X (gfx1250) — compile-verified, matrix operand order checked by reading, not run on hardware
//
#include <hip/hip_runtime.h>
#include <hip/hip_bf16.h>

// ---------------------------------------------------------------------------
// Multi-head attention, CDNA5 (gfx1250), wave32, bf16 WMMA everywhere.
// BS=2, QLEN=2048, DIM=1024, NHEADS=16, HEAD=64, ROWS = BS*QLEN = 4096.
// ---------------------------------------------------------------------------

#define MHA_BS     2
#define MHA_QLEN   2048
#define MHA_DIM    1024
#define MHA_NH     16
#define MHA_HEAD   64
#define MHA_ROWS   (MHA_BS * MHA_QLEN)

typedef __attribute__((ext_vector_type(16))) __bf16 bf16x16;
typedef __attribute__((ext_vector_type(8)))  __bf16 bf16x8;
typedef __attribute__((ext_vector_type(8)))  float  f32x8;

// Load one WMMA 16x16x32 bf16 operand chunk-pair for this lane.
// Per ISA layout (16-bit A / B^T rows): lanes 0-15 hold K=[0..7] and [16..23];
// lanes 16-31 hold K=[8..15] and [24..31]  (klo = half?8:0, khi = klo+16).
__device__ __forceinline__ bf16x16 ld_op(const __bf16* p, int klo, int khi) {
  bf16x8 lo = *reinterpret_cast<const bf16x8*>(p + klo);
  bf16x8 hi = *reinterpret_cast<const bf16x8*>(p + khi);
  bf16x16 r;
#pragma unroll
  for (int i = 0; i < 8; ++i) { r[i] = lo[i]; r[i + 8] = hi[i]; }
  return r;
}

__device__ __forceinline__ f32x8 wmma_bf16(bf16x16 a, bf16x16 b, f32x8 c) {
  return __builtin_amdgcn_wmma_f32_16x16x32_bf16(
      /*neg_a=*/false, a, /*neg_b=*/false, b,
      /*c_mod=*/(short)0, c, /*reuse_a=*/false, /*reuse_b=*/false);
}

// ---------------------------------------------------------------------------
// fp32 -> bf16 elementwise convert
// ---------------------------------------------------------------------------
__global__ void mha_f32_to_bf16(const float* __restrict__ src,
                                __bf16* __restrict__ dst, int n) {
  int i = blockIdx.x * blockDim.x + threadIdx.x;
  if (i < n) dst[i] = (__bf16)src[i];
}

// ---------------------------------------------------------------------------
// NT GEMM: C[m,n] = (sum_k A[m,k]*B[n,k] + bias[n]) * scale
//   A: [M,K] bf16 row-major, B: [N,K] bf16 row-major (== torch Linear weight)
// mode 0: store bf16 at out[row*N + n]
// mode 1: store bf16 V-transposed: out[((b*16+h)*64 + d)*QLEN + pos]
// mode 2: store fp32 at out[row*N + n]
// Block: 256 thr = 8 waves (2 x 4). Wave tile 64x64 (16 accumulators, 4x A/B
// operand reuse), block tile 128(M) x 256(N).
// ---------------------------------------------------------------------------
__global__ void mha_gemm_nt_bf16(const __bf16* __restrict__ A,
                                 const __bf16* __restrict__ B,
                                 const float* __restrict__ bias,
                                 void* __restrict__ outp,
                                 int M, int N, int K, int mode, float scale) {
  const int lane = threadIdx.x & 31;
  const int wave = threadIdx.x >> 5;
  const int wm = wave & 1, wn = wave >> 1;
  const int m0 = blockIdx.x * 128 + wm * 64;
  const int n0 = blockIdx.y * 256 + wn * 64;
  const int half = lane >> 4, l16 = lane & 15;
  const int klo = half ? 8 : 0, khi = klo + 16;

  const __bf16* arow[4];
  const __bf16* brow[4];
#pragma unroll
  for (int t = 0; t < 4; ++t) {
    arow[t] = A + (size_t)(m0 + t * 16 + l16) * K;
    brow[t] = B + (size_t)(n0 + t * 16 + l16) * K;
  }

  f32x8 acc[4][4];
#pragma unroll
  for (int i = 0; i < 4; ++i)
#pragma unroll
    for (int j = 0; j < 4; ++j) acc[i][j] = f32x8{};

  for (int k = 0; k < K; k += 32) {
    if (k + 64 < K) {  // speculative prefetch two k-steps ahead
      __builtin_prefetch(arow[0] + k + 64, 0, 1);
      __builtin_prefetch(brow[0] + k + 64, 0, 1);
    }
    bf16x16 a[4], b[4];
#pragma unroll
    for (int t = 0; t < 4; ++t) {
      a[t] = ld_op(arow[t] + k, klo, khi);
      b[t] = ld_op(brow[t] + k, klo, khi);
    }
#pragma unroll
    for (int i = 0; i < 4; ++i)
#pragma unroll
      for (int j = 0; j < 4; ++j)
        acc[i][j] = wmma_bf16(a[i], b[j], acc[i][j]);
  }

  // C layout: vgpr e, lanes 0-15 -> (M=e, N=l16); lanes 16-31 -> (M=e+8, N=l16)
#pragma unroll
  for (int ti = 0; ti < 4; ++ti) {
#pragma unroll
    for (int tj = 0; tj < 4; ++tj) {
#pragma unroll
      for (int e = 0; e < 8; ++e) {
        int row = m0 + ti * 16 + e + 8 * half;
        int n = n0 + tj * 16 + l16;
        float v = (acc[ti][tj][e] + bias[n]) * scale;
        if (mode == 0) {
          ((__bf16*)outp)[(size_t)row * N + n] = (__bf16)v;
        } else if (mode == 1) {
          int b_ = row >> 11, pos = row & (MHA_QLEN - 1);
          int h = n >> 6, d = n & (MHA_HEAD - 1);
          ((__bf16*)outp)[(((size_t)(b_ * MHA_NH + h)) * MHA_HEAD + d) * MHA_QLEN + pos] =
              (__bf16)v;
        } else {
          ((float*)outp)[(size_t)row * N + n] = v;
        }
      }
    }
  }
}

// ---------------------------------------------------------------------------
// Flash attention: per (b,h), streaming online softmax over 64-key blocks.
// Grid: (QLEN/128, BS*NH), block 256 = 8 waves; each wave owns 16 query rows.
// Q,K: bf16 [ROWS, DIM] (Q pre-scaled by 1/sqrt(HEAD) in projection)
// Vt:  bf16 [BS*NH, HEAD, QLEN]  (transposed so P@V B-operand rows are dense)
// ctx: bf16 [ROWS, DIM]
// ---------------------------------------------------------------------------
#define P_STRIDE 72  // 64 + 8 pad, x2B = 144 B row stride (16B-aligned chunks)

__global__ void mha_flash_attn(const __bf16* __restrict__ Q,
                               const __bf16* __restrict__ Km,
                               const __bf16* __restrict__ Vt,
                               const float* __restrict__ mask,
                               __bf16* __restrict__ ctx) {
  __shared__ __align__(16) __bf16 lds_p[8][16 * P_STRIDE];  // per-wave 16x64 P

  const int lane = threadIdx.x & 31;
  const int wave = threadIdx.x >> 5;
  const int half = lane >> 4, l16 = lane & 15;
  const int klo = half ? 8 : 0, khi = klo + 16;

  const int bh = blockIdx.y;
  const int b = bh >> 4, h = bh & (MHA_NH - 1);
  const int q0 = blockIdx.x * 128 + wave * 16;  // query offset within batch

  // Q tile for this wave: 16 rows x HEAD(64) -> two k-step A operands
  const __bf16* qrow =
      Q + (size_t)(b * MHA_QLEN + q0 + l16) * MHA_DIM + h * MHA_HEAD;
  const bf16x16 aq0 = ld_op(qrow, klo, khi);
  const bf16x16 aq1 = ld_op(qrow + 32, klo, khi);

  const __bf16* kbase = Km + (size_t)b * MHA_QLEN * MHA_DIM + h * MHA_HEAD;
  const __bf16* vbase = Vt + (size_t)bh * MHA_HEAD * MHA_QLEN;
  const float* mrow = mask + (size_t)b * MHA_QLEN;

  float m_s[8], l_s[8];
  f32x8 o[4] = {{}, {}, {}, {}};
#pragma unroll
  for (int i = 0; i < 8; ++i) { m_s[i] = -3.0e38f; l_s[i] = 0.0f; }

  __bf16* pl = &lds_p[wave][0];

  for (int j = 0; j < MHA_QLEN; j += 64) {
    // ---- S = Q @ K^T for key block [j, j+64) : four 16x16 tiles ----
    f32x8 s[4];
    float mk[4];
#pragma unroll
    for (int t = 0; t < 4; ++t) {
      const __bf16* kr = kbase + (size_t)(j + t * 16 + l16) * MHA_DIM;
      f32x8 st = {};
      st = wmma_bf16(aq0, ld_op(kr, klo, khi), st);
      st = wmma_bf16(aq1, ld_op(kr + 32, klo, khi), st);
      s[t] = st;
      mk[t] = mrow[j + t * 16 + l16];
    }

    // ---- online softmax, per C-layout row (vgpr i -> row i + 8*half) ----
#pragma unroll
    for (int i = 0; i < 8; ++i) {
      float e0 = s[0][i] + mk[0];
      float e1 = s[1][i] + mk[1];
      float e2 = s[2][i] + mk[2];
      float e3 = s[3][i] + mk[3];
      float v = fmaxf(fmaxf(e0, e1), fmaxf(e2, e3));
      v = fmaxf(v, __shfl_xor(v, 1, 32));
      v = fmaxf(v, __shfl_xor(v, 2, 32));
      v = fmaxf(v, __shfl_xor(v, 4, 32));
      v = fmaxf(v, __shfl_xor(v, 8, 32));  // row max within 16-lane half
      float mnew = fmaxf(m_s[i], v);
      float alpha = __expf(m_s[i] - mnew);
      float p0 = __expf(e0 - mnew);
      float p1 = __expf(e1 - mnew);
      float p2 = __expf(e2 - mnew);
      float p3 = __expf(e3 - mnew);
      float rs = (p0 + p1) + (p2 + p3);
      rs += __shfl_xor(rs, 1, 32);
      rs += __shfl_xor(rs, 2, 32);
      rs += __shfl_xor(rs, 4, 32);
      rs += __shfl_xor(rs, 8, 32);
      l_s[i] = l_s[i] * alpha + rs;
      m_s[i] = mnew;
      o[0][i] *= alpha;
      o[1][i] *= alpha;
      o[2][i] *= alpha;
      o[3][i] *= alpha;
      // spill P (C layout) to LDS as bf16
      int r = i + 8 * half;
      pl[r * P_STRIDE + l16]      = (__bf16)p0;
      pl[r * P_STRIDE + 16 + l16] = (__bf16)p1;
      pl[r * P_STRIDE + 32 + l16] = (__bf16)p2;
      pl[r * P_STRIDE + 48 + l16] = (__bf16)p3;
    }

    // ---- reload P in A-operand layout: two 16x32 operands ----
    const __bf16* prow = pl + l16 * P_STRIDE;
    const bf16x16 pa0 = ld_op(prow, klo, khi);        // keys j   .. j+31
    const bf16x16 pa1 = ld_op(prow + 32, klo, khi);   // keys j+32.. j+63

    // ---- O += P @ V : 4 output tiles over the 64 head dims ----
#pragma unroll
    for (int t = 0; t < 4; ++t) {
      const __bf16* vrow = vbase + (size_t)(t * 16 + l16) * MHA_QLEN + j;
      f32x8 ot = o[t];
      ot = wmma_bf16(pa0, ld_op(vrow, klo, khi), ot);
      ot = wmma_bf16(pa1, ld_op(vrow + 32, klo, khi), ot);
      o[t] = ot;
    }
  }

  // ---- finalize: divide by l, store context bf16 [ROWS, DIM] ----
#pragma unroll
  for (int i = 0; i < 8; ++i) {
    float inv = 1.0f / l_s[i];
    size_t row = (size_t)(b * MHA_QLEN + q0 + i + 8 * half);
#pragma unroll
    for (int t = 0; t < 4; ++t) {
      ctx[row * MHA_DIM + h * MHA_HEAD + t * 16 + l16] = (__bf16)(o[t][i] * inv);
    }
  }
}

// ---------------------------------------------------------------------------
// Host-side orchestration
// ---------------------------------------------------------------------------
extern "C" void kernel_launch(void* const* d_in, const int* in_sizes, int n_in,
                              void* d_out, int out_size, void* d_ws, size_t ws_size,
                              hipStream_t stream) {
  (void)in_sizes; (void)n_in; (void)out_size; (void)ws_size;

  const float* input = (const float*)d_in[0];
  const float* mask  = (const float*)d_in[1];
  const float* q_w   = (const float*)d_in[2];
  const float* q_b   = (const float*)d_in[3];
  const float* k_w   = (const float*)d_in[4];
  const float* k_b   = (const float*)d_in[5];
  const float* v_w   = (const float*)d_in[6];
  const float* v_b   = (const float*)d_in[7];
  const float* o_w   = (const float*)d_in[8];
  const float* o_b   = (const float*)d_in[9];
  float* out = (float*)d_out;

  const size_t szX = (size_t)MHA_ROWS * MHA_DIM * sizeof(__bf16);  // 8 MiB
  const size_t szW = (size_t)MHA_DIM * MHA_DIM * sizeof(__bf16);   // 2 MiB

  char* ws = (char*)d_ws;
  __bf16* xb  = (__bf16*)(ws);
  __bf16* wqb = (__bf16*)(ws + szX);
  __bf16* wkb = (__bf16*)(ws + szX + szW);
  __bf16* wvb = (__bf16*)(ws + szX + 2 * szW);
  __bf16* wob = (__bf16*)(ws + szX + 3 * szW);
  __bf16* qb  = (__bf16*)(ws + szX + 4 * szW);
  __bf16* kb  = (__bf16*)(ws + 2 * szX + 4 * szW);
  __bf16* vt  = (__bf16*)(ws + 3 * szX + 4 * szW);
  __bf16* cb  = (__bf16*)(ws + 4 * szX + 4 * szW);
  // total workspace: 5*8MiB + 4*2MiB ~= 48 MiB

  const int nX = MHA_ROWS * MHA_DIM;
  const int nW = MHA_DIM * MHA_DIM;
  mha_f32_to_bf16<<<(nX + 255) / 256, 256, 0, stream>>>(input, xb, nX);
  mha_f32_to_bf16<<<(nW + 255) / 256, 256, 0, stream>>>(q_w, wqb, nW);
  mha_f32_to_bf16<<<(nW + 255) / 256, 256, 0, stream>>>(k_w, wkb, nW);
  mha_f32_to_bf16<<<(nW + 255) / 256, 256, 0, stream>>>(v_w, wvb, nW);
  mha_f32_to_bf16<<<(nW + 255) / 256, 256, 0, stream>>>(o_w, wob, nW);

  const dim3 gblk(256);
  const dim3 ggrid(MHA_ROWS / 128, MHA_DIM / 256);  // (32, 4)
  const float qscale = 0.125f;  // 1/sqrt(64)

  // Q = (x @ q_w^T + q_b) * qscale   -> bf16 [ROWS, DIM]
  mha_gemm_nt_bf16<<<ggrid, gblk, 0, stream>>>(xb, wqb, q_b, qb,
      MHA_ROWS, MHA_DIM, MHA_DIM, /*mode=*/0, qscale);
  // K = x @ k_w^T + k_b             -> bf16 [ROWS, DIM]
  mha_gemm_nt_bf16<<<ggrid, gblk, 0, stream>>>(xb, wkb, k_b, kb,
      MHA_ROWS, MHA_DIM, MHA_DIM, /*mode=*/0, 1.0f);
  // V = x @ v_w^T + v_b             -> bf16 transposed [BS*NH, HEAD, QLEN]
  mha_gemm_nt_bf16<<<ggrid, gblk, 0, stream>>>(xb, wvb, v_b, vt,
      MHA_ROWS, MHA_DIM, MHA_DIM, /*mode=*/1, 1.0f);

  // flash attention -> ctx bf16 [ROWS, DIM]
  mha_flash_attn<<<dim3(MHA_QLEN / 128, MHA_BS * MHA_NH), gblk, 0, stream>>>(
      qb, kb, vt, mask, cb);

  // out = ctx @ o_w^T + o_b         -> fp32 [ROWS, DIM]
  mha_gemm_nt_bf16<<<ggrid, gblk, 0, stream>>>(cb, wob, o_b, out,
      MHA_ROWS, MHA_DIM, MHA_DIM, /*mode=*/2, 1.0f);
}
